// MeanAggregator_60103772340411
// MI455X (gfx1250) — matrix operand order repaired and checked by reading, not compiled, not run on hardware
//
#include <hip/hip_runtime.h>

// MeanAggregator for MI455X (gfx1250, wave32, WMMA).
//
// out = relu([dif_mat @ feat[src_idx] | feat[dst_idx]] @ w)
//
// GEMM1 (4096x32768x128 = 34.4 GFLOP) streams 512MB of f32 dif_mat: HBM floor
// = 512MB / 23.3 TB/s ~= 23us. Done with v_wmma_f32_16x16x32_bf16 using an
// exact truncation split (a=a_hi+a_lo, b=b_hi+b_lo, keep hi*hi+hi*lo+lo*hi,
// ~2^-16 rel err) so compute sits far above the memory roofline.
//
// Data movement: src rows are gathered+packed once into the per-lane WMMA B
// layout (bf16 hi+lo, 16MB, L2-resident). Each 8-wave workgroup covers 128
// output rows and double-buffers the 16KB/K-block B panel through LDS, so B
// is fetched from L2 only once per workgroup (~512MB L2 total) and A is read
// exactly once from HBM. Each wave owns 2 M-tiles x 4 N-tiles (64 acc VGPRs),
// so every LDS B fragment feeds 6 WMMAs. K is split 8 ways into deterministic
// f32 partial buffers reduced in fixed order by the fused output kernel.

typedef __attribute__((ext_vector_type(16))) __bf16   v16bf;
typedef __attribute__((ext_vector_type(8)))  float    v8f;
typedef __attribute__((ext_vector_type(8)))  unsigned v8u;

#define SRC_DIM 128
#define N_DST   4096
#define N_SRC   32768
#define KBLKS   1024            // 32768 / 32
#define KSPLIT  8
#define KB_PER  (KBLKS / KSPLIT) // 128 K-blocks per workgroup

union BV { uint4 q[2]; v16bf v; };
union PkBf { __bf16 h[2]; unsigned u; };

// ---------------------------------------------------------------------------
// Kernel 1: gather src rows, split to bf16 hi/lo, pack into WMMA-B layout.
// B tile (kb,nt): lane = khalf*16 + n_local, dword j holds K = khalf*16+2j(+1).
// Buffer layout (u32 units): [kb][nt][chunk c=j>>2][lane][jj=j&3]
//   -> compute reads one uint4 per (chunk,lane), stride 16B across lanes
//      (bank-sequential in LDS after staging).
// ---------------------------------------------------------------------------
__global__ __launch_bounds__(256) void pack_b_kernel(
    const float* __restrict__ feat, const int* __restrict__ sidx,
    unsigned* __restrict__ bhi, unsigned* __restrict__ blo) {
  int tid  = blockIdx.x * 256 + threadIdx.x;   // 0 .. 2097151
  int jj   = tid & 3;
  int lane = (tid >> 2) & 31;
  int c    = (tid >> 7) & 1;
  int nt   = (tid >> 8) & 7;
  int kb   = tid >> 11;
  int j    = c * 4 + jj;
  int khalf = lane >> 4;
  int n    = (lane & 15) + nt * 16;
  int k0   = kb * 32 + khalf * 16 + 2 * j;
  int r0 = sidx[k0];
  int r1 = sidx[k0 + 1];
  float v0 = feat[r0 * SRC_DIM + n];
  float v1 = feat[r1 * SRC_DIM + n];
  // exact truncation split: hi = top 16 bits, lo = v - hi (exact in f32)
  unsigned u0 = __builtin_bit_cast(unsigned, v0);
  unsigned u1 = __builtin_bit_cast(unsigned, v1);
  unsigned hpack = (u1 & 0xffff0000u) | (u0 >> 16);
  float h0 = __builtin_bit_cast(float, u0 & 0xffff0000u);
  float h1 = __builtin_bit_cast(float, u1 & 0xffff0000u);
  PkBf pl;
  pl.h[0] = (__bf16)(v0 - h0);
  pl.h[1] = (__bf16)(v1 - h1);
  bhi[tid] = hpack;
  blo[tid] = pl.u;
}

// ---------------------------------------------------------------------------
// Kernel 2: gather dst rows (exact f32 copy).
// ---------------------------------------------------------------------------
__global__ __launch_bounds__(256) void gather_dst_kernel(
    const float* __restrict__ feat, const int* __restrict__ didx,
    float* __restrict__ dstg) {
  int t = blockIdx.x * 256 + threadIdx.x;      // 0 .. 524287
  int row = t >> 7, col = t & 127;
  dstg[t] = feat[didx[row] * SRC_DIM + col];
}

// ---------------------------------------------------------------------------
// Kernel 3: partial[ks] += dif_mat(ks K-range) @ src_packed, WMMA bf16-split.
// Grid: 256 blocks x 256 thr. block -> (mblock = bx&31: 128 rows, ks = bx>>5).
// Wave w: M-tile pair p=w>>1 (rows mblock*128 + p*32 .. +31),
//         N-tile half h=w&1 (cols h*64 .. +63).
// ---------------------------------------------------------------------------
__global__ __launch_bounds__(256) void gemm_agg_kernel(
    const float* __restrict__ A,
    const uint4* __restrict__ Bhi, const uint4* __restrict__ Blo,
    float* __restrict__ partials) {
  __shared__ uint4 sB[2][1024];                 // 2 x (8KB hi + 8KB lo)

  const int tid   = threadIdx.x;
  const int lane  = tid & 31;
  const int w     = tid >> 5;
  const int p     = w >> 1;                     // 0..3 : M-tile pair
  const int h     = w & 1;                      // 0..1 : N-tile half
  const int mblock = blockIdx.x & 31;
  const int ks     = blockIdx.x >> 5;
  const int kb0    = ks * KB_PER;
  const int khalf  = lane >> 4;

  const float* __restrict__ arow0 =
      A + (size_t)((mblock * 8 + 2 * p) * 16 + (lane & 15)) * (size_t)N_SRC;
  const float* __restrict__ arow1 = arow0 + (size_t)16 * N_SRC;
  float* __restrict__ pout = partials + (size_t)ks * (N_DST * SRC_DIM);

  v8f acc[2][4] = {};

  // stage B panel for a K-block into LDS buffer `buf` (16KB, 64B/thread)
  auto stage = [&](int kb, int buf) {
    const uint4* gh = Bhi + (size_t)kb * 512 + tid * 2;
    const uint4* gl = Blo + (size_t)kb * 512 + tid * 2;
    sB[buf][tid * 2]           = gh[0];
    sB[buf][tid * 2 + 1]       = gh[1];
    sB[buf][512 + tid * 2]     = gl[0];
    sB[buf][512 + tid * 2 + 1] = gl[1];
  };

  stage(kb0, 0);
  __syncthreads();

  for (int i = 0; i < KB_PER; ++i) {
    const int kb  = kb0 + i;
    const int cur = i & 1;
    if (i + 1 < KB_PER) stage(kb + 1, cur ^ 1);   // uniform branch

    // ---- A: load 16 f32 per lane per M-tile, truncation-split to bf16 ----
    const int kbase = kb * 32 + khalf * 16;
    __builtin_prefetch(arow0 + kbase + 256, 0, 1);  // 8 K-blocks ahead
    __builtin_prefetch(arow1 + kbase + 256, 0, 1);

    v16bf ahi[2], alo[2];
    const float* ar[2] = {arow0, arow1};
#pragma unroll
    for (int t = 0; t < 2; ++t) {
      float4 a0 = *(const float4*)(ar[t] + kbase);
      float4 a1 = *(const float4*)(ar[t] + kbase + 4);
      float4 a2 = *(const float4*)(ar[t] + kbase + 8);
      float4 a3 = *(const float4*)(ar[t] + kbase + 12);
      float av[16] = {a0.x, a0.y, a0.z, a0.w, a1.x, a1.y, a1.z, a1.w,
                      a2.x, a2.y, a2.z, a2.w, a3.x, a3.y, a3.z, a3.w};
      v8u hb;
#pragma unroll
      for (int jp = 0; jp < 8; ++jp) {
        unsigned u0 = __builtin_bit_cast(unsigned, av[2 * jp]);
        unsigned u1 = __builtin_bit_cast(unsigned, av[2 * jp + 1]);
        hb[jp] = (u1 & 0xffff0000u) | (u0 >> 16);
        float h0 = __builtin_bit_cast(float, u0 & 0xffff0000u);
        float h1 = __builtin_bit_cast(float, u1 & 0xffff0000u);
        alo[t][2 * jp]     = (__bf16)(av[2 * jp] - h0);
        alo[t][2 * jp + 1] = (__bf16)(av[2 * jp + 1] - h1);
      }
      ahi[t] = __builtin_bit_cast(v16bf, hb);
    }

    // ---- B from LDS: 4 N-tiles, each feeds 2 M-tiles x 3 WMMAs ----
#pragma unroll
    for (int q = 0; q < 4; ++q) {
      const int nt = h * 4 + q;
      BV bh, bl;
      bh.q[0] = sB[cur][nt * 64 + lane];
      bh.q[1] = sB[cur][nt * 64 + 32 + lane];
      bl.q[0] = sB[cur][512 + nt * 64 + lane];
      bl.q[1] = sB[cur][512 + nt * 64 + 32 + lane];
#pragma unroll
      for (int t = 0; t < 2; ++t) {
        v8f c = acc[t][q];
        c = __builtin_amdgcn_wmma_f32_16x16x32_bf16(false, ahi[t], false, bh.v,
                                                    (short)0, c, false, false);
        c = __builtin_amdgcn_wmma_f32_16x16x32_bf16(false, ahi[t], false, bl.v,
                                                    (short)0, c, false, false);
        c = __builtin_amdgcn_wmma_f32_16x16x32_bf16(false, alo[t], false, bh.v,
                                                    (short)0, c, false, false);
        acc[t][q] = c;
      }
    }
    __syncthreads();
  }

  // C/D layout: VGPR r -> M = khalf*8 + r, N = lane&15 within tile.
#pragma unroll
  for (int t = 0; t < 2; ++t) {
    const int rowb = (mblock * 8 + 2 * p + t) * 16 + khalf * 8;
#pragma unroll
    for (int q = 0; q < 4; ++q) {
      const int col = (h * 4 + q) * 16 + (lane & 15);
#pragma unroll
      for (int r = 0; r < 8; ++r)
        pout[(size_t)(rowb + r) * SRC_DIM + col] = acc[t][q][r];
    }
  }
}

// ---------------------------------------------------------------------------
// Kernel 4: out = relu([sum_ks partial | dstg] @ w), exact f32, fixed-order
// reduction (deterministic). One 128-thread block per dst row.
// ---------------------------------------------------------------------------
__global__ __launch_bounds__(128) void fuse_out_kernel(
    const float* __restrict__ partials, const float* __restrict__ dstg,
    const float* __restrict__ w, float* __restrict__ out) {
  __shared__ float srow[256];
  const int i = blockIdx.x;
  const int o = threadIdx.x;
  float s = 0.f;
#pragma unroll
  for (int ksp = 0; ksp < KSPLIT; ++ksp)
    s += partials[(size_t)ksp * (N_DST * SRC_DIM) + i * SRC_DIM + o];
  srow[o]       = s;
  srow[128 + o] = dstg[i * 128 + o];
  __syncthreads();
  float acc = 0.f;
#pragma unroll 8
  for (int k = 0; k < 256; ++k) acc = fmaf(srow[k], w[k * 128 + o], acc);
  out[i * 128 + o] = fmaxf(acc, 0.f);
}

// ---------------------------------------------------------------------------
extern "C" void kernel_launch(void* const* d_in, const int* in_sizes, int n_in,
                              void* d_out, int out_size, void* d_ws, size_t ws_size,
                              hipStream_t stream) {
  const float* feat = (const float*)d_in[0];   // [36864,128]
  const float* dif  = (const float*)d_in[1];   // [4096,32768]
  const float* w    = (const float*)d_in[2];   // [256,128]
  const int*   didx = (const int*)d_in[3];     // [4096]
  const int*   sidx = (const int*)d_in[4];     // [32768]
  float* out = (float*)d_out;                  // [4096,128]

  char* ws = (char*)d_ws;
  unsigned* bhi      = (unsigned*)(ws);                      // 8 MB
  unsigned* blo      = (unsigned*)(ws + (size_t)( 8 << 20)); // 8 MB
  float*    dstg     = (float*)   (ws + (size_t)(16 << 20)); // 2 MB
  float*    partials = (float*)   (ws + (size_t)(18 << 20)); // 16 MB

  pack_b_kernel<<<8192, 256, 0, stream>>>(feat, sidx, bhi, blo);
  gather_dst_kernel<<<2048, 256, 0, stream>>>(feat, didx, dstg);
  gemm_agg_kernel<<<256, 256, 0, stream>>>(dif, (const uint4*)bhi,
                                           (const uint4*)blo, partials);
  fuse_out_kernel<<<4096, 128, 0, stream>>>(partials, dstg, w, out);
  (void)in_sizes; (void)n_in; (void)out_size; (void)ws_size;
}
